// QuantumHybridNAT_65481071402164
// MI455X (gfx1250) — matrix operand order repaired and checked by reading, not compiled, not run on hardware
//
#include <hip/hip_runtime.h>
#include <hip/hip_bf16.h>

// MI455X / gfx1250, wave32. Fully fused CNN + 4-qubit statevector sim.
// Kernel 1: one wave per image:
//   conv1+relu+pool (register-patch VALU, halo-padded LDS, no branches) ->
//   conv2 as implicit GEMM via v_wmma_f32_16x16x32_f16 with channel-last
//   f16 LDS tiles (B half-groups = single ds_load_b128) ->
//   relu+pool+FC -> quantum circuit in registers -> z[b,4] to ws.
// Kernel 2: batch-wide BatchNorm1d(4) (training mode, biased var).
//
// conv2 GEMM uses permuted reduction index k = tap*8 + c (tap=3x3 position,
// c=input channel); A and B use the same permutation so D is unchanged.

typedef __attribute__((ext_vector_type(16))) _Float16 v16h;
typedef __attribute__((ext_vector_type(8)))  _Float16 v8h;
typedef __attribute__((ext_vector_type(8)))  float    v8f;

#define INV_SQRT2f 0.70710678118654752440f

__device__ __forceinline__ float2 cxshfl_xor(float2 a, int mask) {
  return make_float2(__shfl_xor(a.x, mask, 16), __shfl_xor(a.y, mask, 16));
}

// Generic single-qubit gate on bit `mbit` of 4-bit index `i`.
__device__ __forceinline__ void gate1q(float2& a, int i, int mbit,
                                       float2 u00, float2 u01,
                                       float2 u10, float2 u11) {
  float2 p = cxshfl_xor(a, mbit);
  bool t = (i & mbit) != 0;
  float2 a0 = t ? p : a;
  float2 a1 = t ? a : p;
  float2 u0 = t ? u10 : u00;
  float2 u1 = t ? u11 : u01;
  a.x = u0.x*a0.x - u0.y*a0.y + u1.x*a1.x - u1.y*a1.y;
  a.y = u0.x*a0.y + u0.y*a0.x + u1.x*a1.y + u1.y*a1.x;
}

__global__ __launch_bounds__(64)
void fused_cnn_quantum(const float* __restrict__ x,
                       const float* __restrict__ w1, const float* __restrict__ b1,
                       const float* __restrict__ w2, const float* __restrict__ b2,
                       const float* __restrict__ fcw, const float* __restrict__ fcb,
                       const float* __restrict__ Ure, const float* __restrict__ Uim,
                       const float* __restrict__ rx0, const float* __restrict__ ry0,
                       const float* __restrict__ rz0, const float* __restrict__ crx0,
                       float* __restrict__ zbuf, int B)
{
  // Per-image LDS (2 waves / block, one image per wave): ~41 KB total
  __shared__ float    s_inp[2][900];             // 30x30 halo-padded input
  __shared__ __align__(16) _Float16 s_f1c[2][2048]; // feat1 halo: [16][16][8ch]
  __shared__ float    s_c2[2][3136];             // conv2 pre-pool: 16x14x14
  __shared__ float    s_w1[2][72];
  __shared__ float    s_b1[2][8];

  const int lane = threadIdx.x & 31;
  const int wv   = threadIdx.x >> 5;
  const int img  = blockIdx.x * 2 + wv;
  const bool active = img < B;

  // ---- stage conv1 weights; zero halo buffers ------------------------------
  for (int t = lane; t < 72; t += 32) s_w1[wv][t] = w1[t];
  if (lane < 8) s_b1[wv][lane] = b1[lane];
  for (int t = lane; t < 900; t += 32) s_inp[wv][t] = 0.f;
  {
    uint4 z4 = make_uint4(0u, 0u, 0u, 0u);
    uint4* f1z = (uint4*)&s_f1c[wv][0];
    for (int t = lane; t < 256; t += 32) f1z[t] = z4;   // 2048 f16 = 256x16B
  }

  // ---- load input image into halo interior (float4, rows of 28 = 7x4) -----
  if (active) {
    const float4* xi4 = (const float4*)(x + (size_t)img * 784);
    for (int t = lane; t < 196; t += 32) {
      float4 v = xi4[t];
      int row = t / 7, col = (t - row*7) * 4;
      float* dst = &s_inp[wv][(row + 1)*30 + col + 1];
      dst[0] = v.x; dst[1] = v.y; dst[2] = v.z; dst[3] = v.w;
    }
  }

  // ---- conv2 A fragments: 16x96 f16, permuted k = tap*8 + c ----------------
  // A layout (16-bit 16x32): lane group g, row m=lane&15,
// half h -> k = 32q + 16g + h.
  const int mrow = lane & 15;
  const int g    = lane >> 4;
  v16h afr[3];
  #pragma unroll
  for (int q = 0; q < 3; ++q) {
    #pragma unroll
    for (int h = 0; h < 16; ++h) {
      int k   = 32*q + 16*g + h;
      int c   = k & 7;          // input channel
      int tap = k >> 3;         // 3x3 position (valid < 9)
      float wval = (tap < 9) ? w2[mrow*72 + c*9 + tap] : 0.f;
      afr[q][h] = (_Float16)wval;
    }
  }
  float cbias[8];
  #pragma unroll
  for (int r = 0; r < 8; ++r) cbias[r] = b2[r + 8*g];

  // Per-lane tap offsets for B fragments (depend only on g): u = 2q + (hi?1:0)
  int  offs[6];
  bool tapv[6];
  #pragma unroll
  for (int u = 0; u < 6; ++u) {
    int tap  = (u >> 1)*4 + 2*g + (u & 1);
    bool vld = tap < 9;
    int t    = vld ? tap : 0;
    offs[u]  = ((t/3)*16 + (t - (t/3)*3)) * 8;  // halo-pos offset, in halves
    tapv[u]  = vld;
  }

  __syncthreads();

  // ---- conv1 (3x3 SAME) + relu + 2x2 maxpool -> s_f1c ----------------------
  for (int o = lane; o < 1568; o += 32) {
    int c  = o / 196; int rem = o - c*196;
    int py = rem / 14; int px = rem - py*14;
    float patch[16];
    #pragma unroll
    for (int iy = 0; iy < 4; ++iy)
      #pragma unroll
      for (int ix = 0; ix < 4; ++ix)
        patch[iy*4 + ix] = s_inp[wv][(2*py + iy)*30 + (2*px + ix)];
    float mx = -3.4e38f;
    #pragma unroll
    for (int dy = 0; dy < 2; ++dy)
    #pragma unroll
    for (int dx = 0; dx < 2; ++dx) {
      float acc = s_b1[wv][c];
      #pragma unroll
      for (int ty = 0; ty < 3; ++ty)
        #pragma unroll
        for (int tx = 0; tx < 3; ++tx)
          acc += s_w1[wv][c*9 + ty*3 + tx] * patch[(dy + ty)*4 + (dx + tx)];
      mx = fmaxf(mx, acc);
    }
    // channel-last halo store: feat1 (py,px) -> halo (py+1, px+1)
    s_f1c[wv][((py + 1)*16 + (px + 1))*8 + c] = (_Float16)fmaxf(mx, 0.f);
  }

  __syncthreads();

  // ---- conv2 via WMMA: D[16x196] = A[16x96] x B[96x196] + bias, relu -------
  // B layout: lane group g holds k rows [16g,16g+16) of each 32-chunk,
  // column n = lane&15, half h -> k = 32q + 16g + h. With k = tap*8 + c,
  // halves 0..7  = channels 0..7 at tap (4q+2g)    -> one ds_load_b128
  // halves 8..15 = channels 0..7 at tap (4q+2g+1)  -> one ds_load_b128
  const int ncol = lane & 15;
  const _Float16* f1 = &s_f1c[wv][0];
  const v8h zv = {};
  for (int tile = 0; tile < 13; ++tile) {
    int p = tile*16 + ncol;
    bool pv = p < 196;
    int yo = p / 14, xo = p - yo*14;
    if (yo > 13) yo = 13;                  // clamp: garbage cols never stored
    int base = (yo*16 + xo) * 8;           // halo (yo,xo), in halves

    v16h bfr[3];
    #pragma unroll
    for (int q = 0; q < 3; ++q) {
      v8h lo = *(const v8h*)(f1 + base + offs[2*q]);
      v8h hi = *(const v8h*)(f1 + base + offs[2*q + 1]);
      lo = tapv[2*q]     ? lo : zv;
      hi = tapv[2*q + 1] ? hi : zv;
      bfr[q] = __builtin_shufflevector(lo, hi,
                                       0,1,2,3,4,5,6,7,8,9,10,11,12,13,14,15);
    }
    v8f acc;
    #pragma unroll
    for (int r = 0; r < 8; ++r) acc[r] = cbias[r];
    acc = __builtin_amdgcn_wmma_f32_16x16x32_f16(false, afr[0], false, bfr[0],
                                                 (short)0, acc, false, false);
    acc = __builtin_amdgcn_wmma_f32_16x16x32_f16(false, afr[1], false, bfr[1],
                                                 (short)0, acc, false, false);
    acc = __builtin_amdgcn_wmma_f32_16x16x32_f16(false, afr[2], false, bfr[2],
                                                 (short)0, acc, false, false);
    if (pv) {
      #pragma unroll
      for (int r = 0; r < 8; ++r)
        s_c2[wv][(r + 8*g)*196 + p] = fmaxf(acc[r], 0.f);
    }
  }

  __syncthreads();

  // ---- 2x2 maxpool -> [16][7][7], FC 784->4 --------------------------------
  float fsum[4] = {0.f, 0.f, 0.f, 0.f};
  for (int i = lane; i < 784; i += 32) {
    int c  = i / 49; int rem = i - c*49;
    int py = rem / 7; int px = rem - py*7;
    int base2 = c*196 + (2*py)*14 + 2*px;
    float v = fmaxf(fmaxf(s_c2[wv][base2],      s_c2[wv][base2 + 1]),
                    fmaxf(s_c2[wv][base2 + 14], s_c2[wv][base2 + 15]));
    #pragma unroll
    for (int j = 0; j < 4; ++j) fsum[j] += fcw[j*784 + i] * v;
  }
  #pragma unroll
  for (int j = 0; j < 4; ++j) {
    #pragma unroll
    for (int off = 16; off >= 1; off >>= 1)
      fsum[j] += __shfl_xor(fsum[j], off, 32);
    fsum[j] += fcb[j];
  }

  // ---- 4-qubit circuit: amplitude i = lane&15, wire w <-> bit (8 >> w) -----
  const int qi = lane & 15;
  float2 a = make_float2(qi == 0 ? 1.f : 0.f, 0.f);
  const float2 z2c = make_float2(0.f, 0.f);

  { // ry(enc0) wire 0
    float c = cosf(0.5f*fsum[0]), s = sinf(0.5f*fsum[0]);
    gate1q(a, qi, 8, make_float2(c,0), make_float2(-s,0),
                     make_float2(s,0), make_float2(c,0));
  }
  { // rz(enc1) wire 1
    float c = cosf(0.5f*fsum[1]), s = sinf(0.5f*fsum[1]);
    gate1q(a, qi, 4, make_float2(c,-s), z2c, z2c, make_float2(c,s));
  }
  { // rx(enc2) wire 2
    float c = cosf(0.5f*fsum[2]), s = sinf(0.5f*fsum[2]);
    gate1q(a, qi, 2, make_float2(c,0), make_float2(0,-s),
                     make_float2(0,-s), make_float2(c,0));
  }
  { // ry(enc3) wire 3
    float c = cosf(0.5f*fsum[3]), s = sinf(0.5f*fsum[3]);
    gate1q(a, qi, 1, make_float2(c,0), make_float2(-s,0),
                     make_float2(s,0), make_float2(c,0));
  }

  { // 16x16 Haar unitary: a'[i] = sum_j U[i][j] a[j]
    float nr = 0.f, ni = 0.f;
    #pragma unroll
    for (int j = 0; j < 16; ++j) {
      float ajr = __shfl(a.x, j, 16);
      float aji = __shfl(a.y, j, 16);
      float ur = Ure[qi*16 + j], ui = Uim[qi*16 + j];
      nr += ur*ajr - ui*aji;
      ni += ur*aji + ui*ajr;
    }
    a = make_float2(nr, ni);
  }

  { // rx(rx0) wire 0
    float th = rx0[0]; float c = cosf(0.5f*th), s = sinf(0.5f*th);
    gate1q(a, qi, 8, make_float2(c,0), make_float2(0,-s),
                     make_float2(0,-s), make_float2(c,0));
  }
  { // ry(ry0) wire 1
    float th = ry0[0]; float c = cosf(0.5f*th), s = sinf(0.5f*th);
    gate1q(a, qi, 4, make_float2(c,0), make_float2(-s,0),
                     make_float2(s,0), make_float2(c,0));
  }
  { // rz(rz0) wire 3
    float th = rz0[0]; float c = cosf(0.5f*th), s = sinf(0.5f*th);
    gate1q(a, qi, 1, make_float2(c,-s), z2c, z2c, make_float2(c,s));
  }
  { // CRX on [0,2]: control bit 8, rx on bit 2
    float th = crx0[0]; float c = cosf(0.5f*th), s = sinf(0.5f*th);
    float2 p = cxshfl_xor(a, 2);
    if (qi & 8) a = make_float2(c*a.x + s*p.y, c*a.y - s*p.x);
  }
  // H on wire 3 (bit 1)
  gate1q(a, qi, 1, make_float2(INV_SQRT2f,0), make_float2(INV_SQRT2f,0),
                   make_float2(INV_SQRT2f,0), make_float2(-INV_SQRT2f,0));
  // SX on wire 2 (bit 2)
  gate1q(a, qi, 2, make_float2(0.5f,0.5f), make_float2(0.5f,-0.5f),
                   make_float2(0.5f,-0.5f), make_float2(0.5f,0.5f));
  { // CNOT [3,0]: control bit 1, swap over bit 8
    float2 p = cxshfl_xor(a, 8);
    if (qi & 1) a = p;
  }

  // ---- Pauli-Z expectations ------------------------------------------------
  float pr = a.x*a.x + a.y*a.y;
  float e0 = (qi & 8) ? -pr : pr;
  float e1 = (qi & 4) ? -pr : pr;
  float e2 = (qi & 2) ? -pr : pr;
  float e3 = (qi & 1) ? -pr : pr;
  #pragma unroll
  for (int m = 8; m >= 1; m >>= 1) {
    e0 += __shfl_xor(e0, m, 16);
    e1 += __shfl_xor(e1, m, 16);
    e2 += __shfl_xor(e2, m, 16);
    e3 += __shfl_xor(e3, m, 16);
  }
  if (active && lane == 0) {
    float* zp = zbuf + (size_t)img * 4;
    zp[0] = e0; zp[1] = e1; zp[2] = e2; zp[3] = e3;
  }
}

// ---- BatchNorm1d(4), training mode (batch mean, biased var) ----------------
__global__ __launch_bounds__(256)
void bn_kernel(const float* __restrict__ z, const float* __restrict__ gamma,
               const float* __restrict__ beta, float* __restrict__ out, int B)
{
  __shared__ float red[8][256];
  const int tid = threadIdx.x;
  float s[4] = {0,0,0,0}, q[4] = {0,0,0,0};
  for (int r = tid; r < B; r += 256) {
    #pragma unroll
    for (int j = 0; j < 4; ++j) {
      float v = z[r*4 + j];
      s[j] += v; q[j] += v*v;
    }
  }
  #pragma unroll
  for (int j = 0; j < 4; ++j) { red[j][tid] = s[j]; red[4+j][tid] = q[j]; }
  __syncthreads();
  for (int off = 128; off > 0; off >>= 1) {
    if (tid < off) {
      #pragma unroll
      for (int j = 0; j < 8; ++j) red[j][tid] += red[j][tid + off];
    }
    __syncthreads();
  }
  __shared__ float meanv[4], scalev[4], betav[4];
  if (tid < 4) {
    float m   = red[tid][0] / (float)B;
    float var = red[4+tid][0] / (float)B - m*m;
    meanv[tid]  = m;
    scalev[tid] = gamma[tid] / sqrtf(var + 1e-5f);
    betav[tid]  = beta[tid];
  }
  __syncthreads();
  for (int r = tid; r < B; r += 256) {
    #pragma unroll
    for (int j = 0; j < 4; ++j)
      out[r*4 + j] = (z[r*4 + j] - meanv[j]) * scalev[j] + betav[j];
  }
}

extern "C" void kernel_launch(void* const* d_in, const int* in_sizes, int n_in,
                              void* d_out, int out_size, void* d_ws, size_t ws_size,
                              hipStream_t stream) {
  const float* x    = (const float*)d_in[0];
  const float* w1   = (const float*)d_in[1];
  const float* b1   = (const float*)d_in[2];
  const float* w2   = (const float*)d_in[3];
  const float* b2   = (const float*)d_in[4];
  const float* fcw  = (const float*)d_in[5];
  const float* fcb  = (const float*)d_in[6];
  const float* Ure  = (const float*)d_in[7];
  const float* Uim  = (const float*)d_in[8];
  const float* rx0  = (const float*)d_in[9];
  const float* ry0  = (const float*)d_in[10];
  const float* rz0  = (const float*)d_in[11];
  const float* crx0 = (const float*)d_in[12];
  const float* gam  = (const float*)d_in[13];
  const float* bet  = (const float*)d_in[14];
  (void)n_in; (void)out_size; (void)ws_size;

  const int B = in_sizes[0] / 784;          // [B,1,28,28]
  float* zbuf = (float*)d_ws;               // [B,4] pre-BN z values

  dim3 grid((B + 1) / 2);                   // 2 images (waves) per block
  fused_cnn_quantum<<<grid, 64, 0, stream>>>(x, w1, b1, w2, b2, fcw, fcb,
                                             Ure, Uim, rx0, ry0, rz0, crx0,
                                             zbuf, B);
  bn_kernel<<<1, 256, 0, stream>>>(zbuf, gam, bet, (float*)d_out, B);
}
